// HuangM1_59167469469714
// MI455X (gfx1250) — matrix operand order repaired and checked by reading, not compiled, not run on hardware
//
#include <hip/hip_runtime.h>
#include <hip/hip_bf16.h>

#define SEQ     2048
#define DMODEL  1024
#define NHEADS  16
#define HDIM    64
#define BATCH   2
#define CLIPV   16

typedef __attribute__((ext_vector_type(16))) __bf16 v16bf;
typedef __attribute__((ext_vector_type(8)))  float  v8f;
typedef unsigned int __attribute__((ext_vector_type(4))) v4u;
typedef int          __attribute__((ext_vector_type(8))) v8i;
typedef int          __attribute__((ext_vector_type(4))) v4i;

union AFrag { v16bf v; __bf16 e[16]; };
union CFrag { v8f   v; float  e[8];  };

__device__ __forceinline__ v8f wmma_bf16(v16bf a, v16bf b, v8f c) {
    // D = A(16x32 bf16) * B(32x16 bf16) + C(16x16 f32)
    return __builtin_amdgcn_wmma_f32_16x16x32_bf16(
        false, a, false, b, (short)0, c, false, false);
}

// ---------------------------------------------------------------------------
// TDM 2D tile load: global bf16 tensor tile -> LDS, with hardware padding of
// 4 DWORDs every 32 DWORDs (row of 64 bf16 = 128B) => LDS row stride 72 bf16.
// Descriptor packed per CDNA5 ISA 8.3/8.4 (D# group0/group1); groups 2/3 and
// the trailing group are zero (2D tensor). 6-arg builtin form (clang-23).
// ---------------------------------------------------------------------------
__device__ __forceinline__ void tdm_load_2d_bf16(const __bf16* gsrc, void* ldst,
                                                 unsigned tile_w, unsigned tile_h,
                                                 unsigned row_stride_elems)
{
    unsigned long long ga = (unsigned long long)(const void*)gsrc;
    unsigned lds_off = (unsigned)(unsigned long long)ldst; // LDS byte offset (addr[31:0])

    v4u g0;
    g0[0] = 1u;                                     // count=1 (valid user descriptor)
    g0[1] = lds_off;                                // lds_addr [63:32]
    g0[2] = (unsigned)(ga & 0xFFFFFFFFu);           // global_addr[31:0]
    g0[3] = (unsigned)((ga >> 32) & 0x01FFFFFFu)    // global_addr[56:32]
          | 0x80000000u;                            // type=2 ("image") [127:126]

    // group1: data_size=1 (2B); pad_enable, pad_interval=4 (32 DWORDs=128B),
    // pad_amount=3 (4 DWORDs=16B) -> effective LDS row stride 144B = 72 bf16.
    unsigned d0 = (1u << 16) | (1u << 20) | (4u << 22) | (3u << 25);
    unsigned long long td0 = tile_w;                // tensor_dim0 == tile: no OOB
    unsigned long long td1 = tile_h;
    unsigned long long s0  = row_stride_elems;      // tensor_dim0_stride (elems)
    v8i g1;
    g1[0] = (int)d0;
    g1[1] = (int)((td0 & 0xFFFFu) << 16);                            // dim0[15:0] @48
    g1[2] = (int)(((td0 >> 16) & 0xFFFFu) | ((td1 & 0xFFFFu) << 16));// dim0[31:16],dim1[15:0]
    g1[3] = (int)(((td1 >> 16) & 0xFFFFu) | (tile_w << 16));         // dim1[31:16],tile_dim0
    g1[4] = (int)(tile_h & 0xFFFFu);                                 // tile_dim1; tile_dim2=0
    g1[5] = (int)(s0 & 0xFFFFFFFFu);                                 // stride0[31:0]
    g1[6] = (int)((s0 >> 32) & 0xFFFFu);                             // stride0[47:32]
    g1[7] = 0;
    v4i z4 = {0, 0, 0, 0};
    v8i z8 = {0, 0, 0, 0, 0, 0, 0, 0};
    __builtin_amdgcn_tensor_load_to_lds(g0, g1, z4, z4, z8, 0);
}

// ---------------------------------------------------------------------------
// GEMM: C[M,N] = A[M,K](fp32) @ W[K,N](fp32) (+bias), bf16 WMMA inner product.
// 256 threads = 8 waves, 128x128 tile, K-step 32. OUT_BF16 selects epilogue.
// ---------------------------------------------------------------------------
template<bool OUT_BF16>
__global__ __launch_bounds__(256)
void gemm_bf16_kernel(const float* __restrict__ A, const float* __restrict__ W,
                      const float* __restrict__ bias, void* __restrict__ Cout,
                      int M, int N, int K)
{
    __shared__ __bf16 Alds[128][40];   // [m][k]
    __shared__ __bf16 Wt  [128][40];   // [n][k] (W transposed)

    const int t    = threadIdx.x;
    const int wave = t >> 5;
    const int lane = t & 31;
    const int ln   = lane & 15;
    const int hh   = lane >> 4;
    const int m0   = blockIdx.x * 128;
    const int n0   = blockIdx.y * 128;

    CFrag acc[8];
#pragma unroll
    for (int nt = 0; nt < 8; nt++)
#pragma unroll
        for (int i = 0; i < 8; i++) acc[nt].e[i] = 0.0f;

    for (int kb = 0; kb < K; kb += 32) {
#pragma unroll
        for (int it = 0; it < 4; it++) {
            int f = t + 256 * it, r = f >> 3, c4 = f & 7;
            const float4 av = *(const float4*)(A + (size_t)(m0 + r) * K + kb + 4 * c4);
            __bf16* dst = &Alds[r][4 * c4];
            dst[0] = (__bf16)av.x; dst[1] = (__bf16)av.y;
            dst[2] = (__bf16)av.z; dst[3] = (__bf16)av.w;
        }
#pragma unroll
        for (int it = 0; it < 4; it++) {
            int f = t + 256 * it, r = f >> 5, c4 = f & 31;
            const float4 wv = *(const float4*)(W + (size_t)(kb + r) * N + n0 + 4 * c4);
            Wt[4 * c4 + 0][r] = (__bf16)wv.x;
            Wt[4 * c4 + 1][r] = (__bf16)wv.y;
            Wt[4 * c4 + 2][r] = (__bf16)wv.z;
            Wt[4 * c4 + 3][r] = (__bf16)wv.w;
        }
        __syncthreads();

        // Cache-prefetch the next K-step's tiles while we compute this one.
        if (kb + 32 < K) {
            if (t < 128)
                __builtin_prefetch(A + (size_t)(m0 + t) * K + kb + 32, 0, 1);
            else if (t < 160)
                __builtin_prefetch(W + (size_t)(kb + 32 + (t - 128)) * N + n0, 0, 1);
        }

        AFrag a;   // A frag: row m=16w+ln; k=e+16h (contiguous)
#pragma unroll
        for (int e = 0; e < 16; e++) {
            int kk = e + 16 * hh;
            a.e[e] = Alds[16 * wave + ln][kk];
        }
#pragma unroll
        for (int nt = 0; nt < 8; nt++) {
            AFrag b;   // B frag: n=ln+16nt, k=e+16h (contiguous)
#pragma unroll
            for (int e = 0; e < 16; e++) b.e[e] = Wt[ln + 16 * nt][e + 16 * hh];
            acc[nt].v = wmma_bf16(a.v, b.v, acc[nt].v);
        }
        __syncthreads();
    }

#pragma unroll
    for (int nt = 0; nt < 8; nt++)
#pragma unroll
        for (int i = 0; i < 8; i++) {
            int row = m0 + 16 * wave + i + 8 * hh;
            int col = n0 + ln + 16 * nt;
            float val = acc[nt].e[i];
            if (bias) val += bias[col];
            if (OUT_BF16) ((__bf16*)Cout)[(size_t)row * N + col] = (__bf16)val;
            else          ((float*)Cout)[(size_t)row * N + col] = val;
        }
}

// ---------------------------------------------------------------------------
// Flash attention, Q/K/V in bf16. Grid (SEQ/64, BATCH*NHEADS), 128 thr/4 waves.
// K/V tiles double-buffered via TDM tensor_load_to_lds (async with compute).
// ---------------------------------------------------------------------------
__global__ __launch_bounds__(128)
void attn_kernel(const __bf16* __restrict__ Q, const __bf16* __restrict__ Km,
                 const __bf16* __restrict__ Vm, const float* __restrict__ rel,
                 float* __restrict__ ctx)
{
    __shared__ __bf16 Qlds[64][72];        // TDM-padded stride
    __shared__ __bf16 Kl[2][32][72];       // [buf][key][hd]
    __shared__ __bf16 Vl[2][32][72];       // [buf][key][hd]
    __shared__ __bf16 Plds[4][16][40];     // per-wave probs staging
    __shared__ float  relS[CLIPV + 1];     // rel_emb pre-scaled by 1/sqrt(hd)

    const int t    = threadIdx.x;
    const int wave = t >> 5;
    const int lane = t & 31;
    const int ln   = lane & 15;
    const int hh   = lane >> 4;
    const int b    = blockIdx.y / NHEADS;
    const int head = blockIdx.y % NHEADS;
    const int qbase = blockIdx.x * 64;

    const __bf16* Qg = Q  + ((size_t)b * SEQ + qbase) * DMODEL + head * HDIM;
    const __bf16* Kg = Km + (size_t)b * SEQ * DMODEL + head * HDIM;
    const __bf16* Vg = Vm + (size_t)b * SEQ * DMODEL + head * HDIM;

    if (t <= CLIPV) relS[t] = rel[t] * 0.125f;   // fold softmax 1/sqrt(64)

    const int qw    = qbase + 16 * wave;
    const int ktmax = (qbase + 63) >> 5;

    // Prologue: DMA Q tile + first K/V tiles, then handshake via barrier.
    if (wave == 0) {
        tdm_load_2d_bf16(Qg, &Qlds[0][0], HDIM, 64, DMODEL);
        tdm_load_2d_bf16(Kg, &Kl[0][0][0], HDIM, 32, DMODEL);
        tdm_load_2d_bf16(Vg, &Vl[0][0][0], HDIM, 32, DMODEL);
        __builtin_amdgcn_s_wait_tensorcnt(0);
    }
    __syncthreads();

    AFrag qa[2];
#pragma unroll
    for (int ks = 0; ks < 2; ks++)
#pragma unroll
        for (int e = 0; e < 16; e++) {
            int kk = 32 * ks + e + 16 * hh;
            qa[ks].e[e] = Qlds[16 * wave + ln][kk];
        }

    CFrag o[4];
#pragma unroll
    for (int nt = 0; nt < 4; nt++)
#pragma unroll
        for (int i = 0; i < 8; i++) o[nt].e[i] = 0.0f;
    float mrow[8], lrow[8];
#pragma unroll
    for (int i = 0; i < 8; i++) { mrow[i] = -1e30f; lrow[i] = 0.0f; }

    int cur = 0;
    for (int kt = 0; kt <= ktmax; kt++) {
        const int kbase = kt * 32;

        // Kick off next tile's DMA into the other buffer (overlaps compute).
        if (wave == 0 && kt < ktmax) {
            tdm_load_2d_bf16(Kg + (size_t)(kbase + 32) * DMODEL, &Kl[cur ^ 1][0][0], HDIM, 32, DMODEL);
            tdm_load_2d_bf16(Vg + (size_t)(kbase + 32) * DMODEL, &Vl[cur ^ 1][0][0], HDIM, 32, DMODEL);
        }

        if (kbase <= qw + 15) {            // wave-uniform causal prune
            const __bf16 (*Kc)[72] = Kl[cur];
            const __bf16 (*Vc)[72] = Vl[cur];

            CFrag s0, s1;
#pragma unroll
            for (int i = 0; i < 8; i++) { s0.e[i] = 0.0f; s1.e[i] = 0.0f; }
#pragma unroll
            for (int ks = 0; ks < 2; ks++) {
                AFrag bk0, bk1;            // B: n=key, k=hd=e+16h+32ks (contiguous)
#pragma unroll
                for (int e = 0; e < 16; e++) {
                    int hd = e + 16 * hh + 32 * ks;
                    bk0.e[e] = Kc[ln][hd];
                    bk1.e[e] = Kc[ln + 16][hd];
                }
                s0.v = wmma_bf16(qa[ks].v, bk0.v, s0.v);
                s1.v = wmma_bf16(qa[ks].v, bk1.v, s1.v);
            }

            // Rel-bias, causal mask, online softmax (C-layout rows i+8h).
            // Index clamped to [0,16] so relS load is unconditional -> the
            // causal mask becomes a branchless v_cndmask, not an EXEC branch.
#pragma unroll
            for (int i = 0; i < 8; i++) {
                int row = qw + i + 8 * hh;
                int dd0 = row - (kbase + ln);        // >=0 iff j0 unmasked
                int dd1 = dd0 - 16;                  // >=0 iff j1 unmasked
                int c0 = dd0 < 0 ? 0 : (dd0 > CLIPV ? CLIPV : dd0);
                int c1 = dd1 < 0 ? 0 : (dd1 > CLIPV ? CLIPV : dd1);
                float r0 = relS[c0];
                float r1 = relS[c1];
                float v0 = (dd0 >= 0) ? s0.e[i] * r0 : -1e30f;
                float v1 = (dd1 >= 0) ? s1.e[i] * r1 : -1e30f;
                float mx = fmaxf(v0, v1);
                mx = fmaxf(mx, __shfl_xor(mx, 1, 32));
                mx = fmaxf(mx, __shfl_xor(mx, 2, 32));
                mx = fmaxf(mx, __shfl_xor(mx, 4, 32));
                mx = fmaxf(mx, __shfl_xor(mx, 8, 32));
                float mnew = fmaxf(mrow[i], mx);
                float p0 = __expf(v0 - mnew);
                float p1 = __expf(v1 - mnew);
                float ps = p0 + p1;
                ps += __shfl_xor(ps, 1, 32);
                ps += __shfl_xor(ps, 2, 32);
                ps += __shfl_xor(ps, 4, 32);
                ps += __shfl_xor(ps, 8, 32);
                float scl = __expf(mrow[i] - mnew);
                lrow[i] = lrow[i] * scl + ps;
                mrow[i] = mnew;
#pragma unroll
                for (int nt = 0; nt < 4; nt++) o[nt].e[i] *= scl;
                Plds[wave][i + 8 * hh][ln]      = (__bf16)p0;
                Plds[wave][i + 8 * hh][ln + 16] = (__bf16)p1;
            }

            AFrag pa;                      // C->A relayout via per-wave LDS
#pragma unroll
            for (int e = 0; e < 16; e++) {
                int kk = e + 16 * hh;
                pa.e[e] = Plds[wave][ln][kk];
            }
#pragma unroll
            for (int nt = 0; nt < 4; nt++) {
                AFrag vb;                  // B: n=hd (ln+16nt), k=key=e+16h
#pragma unroll
                for (int e = 0; e < 16; e++) vb.e[e] = Vc[e + 16 * hh][ln + 16 * nt];
                o[nt].v = wmma_bf16(pa.v, vb.v, o[nt].v);
            }
        }

        // Close the pipeline stage: DMA done (wave0), then block-wide barrier.
        if (wave == 0) __builtin_amdgcn_s_wait_tensorcnt(0);
        __syncthreads();
        cur ^= 1;
    }

    float* Og = ctx + ((size_t)b * SEQ + qbase) * DMODEL + head * HDIM;
#pragma unroll
    for (int i = 0; i < 8; i++) {
        const float invl = 1.0f / lrow[i];
        const int r = 16 * wave + i + 8 * hh;
#pragma unroll
        for (int nt = 0; nt < 4; nt++) {
            int c = ln + 16 * nt;
            Og[(size_t)r * DMODEL + c] = o[nt].e[i] * invl;
        }
    }
}

// ---------------------------------------------------------------------------
extern "C" void kernel_launch(void* const* d_in, const int* in_sizes, int n_in,
                              void* d_out, int out_size, void* d_ws, size_t ws_size,
                              hipStream_t stream) {
    (void)in_sizes; (void)n_in; (void)out_size; (void)ws_size;
    const float* x   = (const float*)d_in[0];
    const float* Wq  = (const float*)d_in[1];
    const float* Wk  = (const float*)d_in[2];
    const float* Wv  = (const float*)d_in[3];
    const float* Wo  = (const float*)d_in[4];
    const float* bo  = (const float*)d_in[5];
    const float* rel = (const float*)d_in[6];

    const int M = BATCH * SEQ;             // 4096
    const size_t SZ = (size_t)M * DMODEL;  // 4M elems
    __bf16* Qb = (__bf16*)d_ws;            // 8 MB
    __bf16* Kb = Qb + SZ;                  // 8 MB
    __bf16* Vb = Kb + SZ;                  // 8 MB
    float*  Cx = (float*)(Vb + SZ);        // 16 MB  (total ws: 40 MB)

    dim3 gg(M / 128, DMODEL / 128, 1);
    gemm_bf16_kernel<true><<<gg, 256, 0, stream>>>(x, Wq, nullptr, Qb, M, DMODEL, DMODEL);
    gemm_bf16_kernel<true><<<gg, 256, 0, stream>>>(x, Wk, nullptr, Kb, M, DMODEL, DMODEL);
    gemm_bf16_kernel<true><<<gg, 256, 0, stream>>>(x, Wv, nullptr, Vb, M, DMODEL, DMODEL);

    attn_kernel<<<dim3(SEQ / 64, BATCH * NHEADS), 128, 0, stream>>>(Qb, Kb, Vb, rel, Cx);

    gemm_bf16_kernel<false><<<gg, 256, 0, stream>>>(Cx, Wo, bo, d_out, M, DMODEL, DMODEL);
}
